// GAT_21320217657467
// MI455X (gfx1250) — matrix operand order repaired and checked by reading, not compiled, not run on hardware
//
#include <hip/hip_runtime.h>
#include <hip/hip_bf16.h>

typedef _Float16 half_t;
typedef __attribute__((ext_vector_type(16))) _Float16 v16h;
typedef __attribute__((ext_vector_type(8)))  _Float16 v8h;
typedef __attribute__((ext_vector_type(8)))  float    v8f;

#define NODES 50000
#define FIN   128
#define HEADS 8
#define HC    512          // HEADS * 64
#define NCLS  40
#define NCLSP 48           // padded to 3 x 16 WMMA tiles
#define NEG_SLOPE 0.2f

// ---------- monotone float <-> uint mapping for atomic max over floats ----------
__device__ __forceinline__ unsigned fenc(float f) {
    unsigned u = __float_as_uint(f);
    return (u >> 31) ? ~u : (u | 0x80000000u);
}
__device__ __forceinline__ float fdec(unsigned u) {
    return (u >> 31) ? __uint_as_float(u ^ 0x80000000u) : __uint_as_float(~u);
}
__device__ __forceinline__ float lrelu(float t) { return t > 0.f ? t : NEG_SLOPE * t; }

// ---------------------------- conversions ----------------------------
__global__ void k_cvt_x(const float* __restrict__ x, half_t* __restrict__ xf, int n) {
    int i = blockIdx.x * 256 + threadIdx.x;
    if (i < n) xf[i] = (half_t)x[i];
}

// W1 [128,512] -> W1t f16 [512,128]  (K contiguous per output column)
__global__ void k_cvt_w1t(const float* __restrict__ W1, half_t* __restrict__ W1t) {
    int i = blockIdx.x * 256 + threadIdx.x;
    if (i >= FIN * HC) return;
    int k = i / HC, n = i - k * HC;
    W1t[n * FIN + k] = (half_t)W1[i];
}

// W2 [512,40] -> W2t f16 [48,512], rows 40..47 zero
__global__ void k_cvt_w2t(const float* __restrict__ W2, half_t* __restrict__ W2t) {
    int i = blockIdx.x * 256 + threadIdx.x;
    if (i >= NCLSP * HC) return;
    int n = i / HC, k = i - n * HC;
    W2t[n * HC + k] = (n < NCLS) ? (half_t)W2[k * NCLS + n] : (half_t)0.0f;
}

// ---------------------------- GEMM1: h1 = x @ W1  (WMMA f16->f32) ----------------------------
// grid = 3125 blocks (16 rows each), block = 256 = 8 waves, wave w owns cols [w*64, w*64+64).
// The 16x128 f16 A tile (4 KB) is shared by all 8 waves: stage it once via
// global_load_async_to_lds_b128 (ASYNCcnt path), then read fragments with ds_load_b128.
__global__ void k_gemm1(const half_t* __restrict__ A, const half_t* __restrict__ Bt,
                        float* __restrict__ C) {
    __shared__ __align__(16) half_t atile[16 * FIN];   // 4 KB
    int tid  = threadIdx.x;
    int wave = tid >> 5, lane = tid & 31;
    int m0 = blockIdx.x * 16;

    // one 16-byte chunk per thread: 256 threads x 16 B = whole 4 KB tile
    {
        const half_t* gsrc = A + (size_t)m0 * FIN + tid * 8;
        unsigned ldsoff = (unsigned)(size_t)(atile + tid * 8);  // low 32 bits = LDS byte addr
        asm volatile("global_load_async_to_lds_b128 %0, %1, off"
                     :: "v"(ldsoff), "v"(gsrc) : "memory");
    }
    asm volatile("s_wait_asynccnt 0x0" ::: "memory");
    __syncthreads();

    int n0 = wave * 64;
    int kgrpA = (lane >> 4) * 16;    // contiguous k: lane half h holds k 16h..16h+15
    int kgrpB = (lane >> 4) * 16;    // B uses the same contiguous k order as A
    int ncol  = lane & 15;
    const half_t* arow = atile + (lane & 15) * FIN;
    v8f acc[4] = {};
    for (int k0 = 0; k0 < FIN; k0 += 32) {
        const half_t* ap = arow + k0 + kgrpA;
        v8h alo = *(const v8h*)ap;          // ds_load_b128
        v8h ahi = *(const v8h*)(ap + 8);    // ds_load_b128
        v16h af;
#pragma unroll
        for (int i = 0; i < 8; ++i) { af[i] = alo[i]; af[i + 8] = ahi[i]; }
#pragma unroll
        for (int nt = 0; nt < 4; ++nt) {
            int n = n0 + nt * 16 + ncol;
            v16h bf = *(const v16h*)(Bt + (size_t)n * FIN + k0 + kgrpB);
            acc[nt] = __builtin_amdgcn_wmma_f32_16x16x32_f16(
                false, af, false, bf, (short)0, acc[nt], false, false);
        }
    }
    int rbase = m0 + ((lane >> 4) << 3);
#pragma unroll
    for (int nt = 0; nt < 4; ++nt) {
        int col = n0 + nt * 16 + ncol;
#pragma unroll
        for (int r = 0; r < 8; ++r)
            C[(size_t)(rbase + r) * HC + col] = acc[nt][r];
    }
}

// ---------------------------- GEMM2: h2 = relu_h @ W2  ----------------------------
// block = 8 waves, each wave owns a private 16-row slab x 48 cols; grid = ceil(50000/128)
__global__ void k_gemm2(const half_t* __restrict__ A, const half_t* __restrict__ Bt,
                        float* __restrict__ C) {
    int wave = threadIdx.x >> 5, lane = threadIdx.x & 31;
    int m0 = blockIdx.x * 128 + wave * 16;
    if (m0 >= NODES) return;                 // wave-uniform: EXEC stays all-ones for WMMA
    int mrow  = m0 + (lane & 15);
    int kgrpA = (lane >> 4) * 16;
    int kgrpB = (lane >> 4) * 16;
    int ncol  = lane & 15;
    v8f acc[3] = {};
    for (int k0 = 0; k0 < HC; k0 += 32) {
        const half_t* ap = A + (size_t)mrow * HC + k0 + kgrpA;
        v8h alo = *(const v8h*)ap;
        v8h ahi = *(const v8h*)(ap + 8);
        v16h af;
#pragma unroll
        for (int i = 0; i < 8; ++i) { af[i] = alo[i]; af[i + 8] = ahi[i]; }
#pragma unroll
        for (int nt = 0; nt < 3; ++nt) {
            v16h bf = *(const v16h*)(Bt + (size_t)(nt * 16 + ncol) * HC + k0 + kgrpB);
            acc[nt] = __builtin_amdgcn_wmma_f32_16x16x32_f16(
                false, af, false, bf, (short)0, acc[nt], false, false);
        }
    }
    int rbase = m0 + ((lane >> 4) << 3);
#pragma unroll
    for (int nt = 0; nt < 3; ++nt) {
        int col = nt * 16 + ncol;
#pragma unroll
        for (int r = 0; r < 8; ++r)
            C[(size_t)(rbase + r) * NCLSP + col] = acc[nt][r];
    }
}

// ---------------------------- attention logits ----------------------------
__global__ void k_attlog1(const float* __restrict__ h, const float* __restrict__ as_,
                          const float* __restrict__ ad_, float* __restrict__ als,
                          float* __restrict__ ald) {
    int idx = blockIdx.x * 256 + threadIdx.x;
    if (idx >= NODES * HEADS) return;
    int n = idx >> 3, hd = idx & 7;
    const float* hp = h + (size_t)n * HC + hd * 64;
    const float* pa = as_ + hd * 64;
    const float* pb = ad_ + hd * 64;
    float s = 0.f, d = 0.f;
#pragma unroll 8
    for (int c = 0; c < 64; ++c) { float v = hp[c]; s += v * pa[c]; d += v * pb[c]; }
    als[idx] = s; ald[idx] = d;
}

__global__ void k_attlog2(const float* __restrict__ h2, const float* __restrict__ as_,
                          const float* __restrict__ ad_, float* __restrict__ als,
                          float* __restrict__ ald) {
    int n = blockIdx.x * 256 + threadIdx.x;
    if (n >= NODES) return;
    const float* hp = h2 + (size_t)n * NCLSP;
    float s = 0.f, d = 0.f;
#pragma unroll 8
    for (int c = 0; c < NCLS; ++c) { float v = hp[c]; s += v * as_[c]; d += v * ad_[c]; }
    als[n] = s; ald[n] = d;
}

// ---------------------------- edge passes, layer 1 (H=8) ----------------------------
__global__ void k_edge_max1(const int* __restrict__ src, const int* __restrict__ dst, int E,
                            int total, const float* __restrict__ als,
                            const float* __restrict__ ald, unsigned* __restrict__ m) {
    int idx = blockIdx.x * 256 + threadIdx.x;
    if (idx >= total) return;
    int e = idx >> 3, hd = idx & 7;
    int s = (e < E) ? src[e] : (e - E);
    int d = (e < E) ? dst[e] : (e - E);
    float t = lrelu(als[s * HEADS + hd] + ald[d * HEADS + hd]);
    atomicMax(&m[d * HEADS + hd], fenc(t));
}

__global__ void k_edge_sum1(const int* __restrict__ src, const int* __restrict__ dst, int E,
                            int total, const float* __restrict__ als,
                            const float* __restrict__ ald, const unsigned* __restrict__ m,
                            float* __restrict__ z) {
    int idx = blockIdx.x * 256 + threadIdx.x;
    if (idx >= total) return;
    int e = idx >> 3, hd = idx & 7;
    int s = (e < E) ? src[e] : (e - E);
    int d = (e < E) ? dst[e] : (e - E);
    float t = lrelu(als[s * HEADS + hd] + ald[d * HEADS + hd]);
    atomicAdd(&z[d * HEADS + hd], expf(t - fdec(m[d * HEADS + hd])));
}

// one block (256 threads) per edge: 512 features, 2 per thread
__global__ void k_scatter1(const int* __restrict__ src, const int* __restrict__ dst, int E,
                           const float* __restrict__ als, const float* __restrict__ ald,
                           const unsigned* __restrict__ m, const float* __restrict__ z,
                           const float* __restrict__ h1, float* __restrict__ agg) {
    int e = blockIdx.x;
    int s = (e < E) ? src[e] : (e - E);
    int d = (e < E) ? dst[e] : (e - E);
    __shared__ float salpha[HEADS];
    int tid = threadIdx.x;
    const float* hs = h1 + (size_t)s * HC;
    // hide the random-row gather latency behind the alpha computation + barrier
    __builtin_prefetch(hs + tid, 0, 0);
    __builtin_prefetch(hs + tid + 256, 0, 0);
    if (tid < HEADS) {
        float t = lrelu(als[s * HEADS + tid] + ald[d * HEADS + tid]);
        salpha[tid] = expf(t - fdec(m[d * HEADS + tid])) / z[d * HEADS + tid];
    }
    __syncthreads();
    float* ad = agg + (size_t)d * HC;
#pragma unroll
    for (int r = 0; r < 2; ++r) {
        int f = tid + r * 256;
        atomicAdd(&ad[f], hs[f] * salpha[f >> 6]);
    }
}

// relu(agg + b1) -> f16 input for layer 2
__global__ void k_relu_cvt(const float* __restrict__ agg, const float* __restrict__ b1,
                           half_t* __restrict__ hf) {
    int i = blockIdx.x * 256 + threadIdx.x;
    if (i >= NODES * HC) return;
    float v = agg[i] + b1[i & (HC - 1)];
    hf[i] = (half_t)(v > 0.f ? v : 0.f);
}

// ---------------------------- edge passes, layer 2 (H=1, C=40) ----------------------------
__global__ void k_edge_max2(const int* __restrict__ src, const int* __restrict__ dst, int E,
                            int Etot, const float* __restrict__ als,
                            const float* __restrict__ ald, unsigned* __restrict__ m) {
    int e = blockIdx.x * 256 + threadIdx.x;
    if (e >= Etot) return;
    int s = (e < E) ? src[e] : (e - E);
    int d = (e < E) ? dst[e] : (e - E);
    atomicMax(&m[d], fenc(lrelu(als[s] + ald[d])));
}

__global__ void k_edge_sum2(const int* __restrict__ src, const int* __restrict__ dst, int E,
                            int Etot, const float* __restrict__ als,
                            const float* __restrict__ ald, const unsigned* __restrict__ m,
                            float* __restrict__ z) {
    int e = blockIdx.x * 256 + threadIdx.x;
    if (e >= Etot) return;
    int s = (e < E) ? src[e] : (e - E);
    int d = (e < E) ? dst[e] : (e - E);
    atomicAdd(&z[d], expf(lrelu(als[s] + ald[d]) - fdec(m[d])));
}

__global__ void k_scatter2(const int* __restrict__ src, const int* __restrict__ dst, int E,
                           int total, const float* __restrict__ als,
                           const float* __restrict__ ald, const unsigned* __restrict__ m,
                           const float* __restrict__ z, const float* __restrict__ h2,
                           float* __restrict__ out) {
    int idx = blockIdx.x * 256 + threadIdx.x;
    if (idx >= total) return;
    int e = idx / NCLS, c = idx - e * NCLS;
    int s = (e < E) ? src[e] : (e - E);
    int d = (e < E) ? dst[e] : (e - E);
    float alpha = expf(lrelu(als[s] + ald[d]) - fdec(m[d])) / z[d];
    atomicAdd(&out[(size_t)d * NCLS + c], h2[(size_t)s * NCLSP + c] * alpha);
}

__global__ void k_bias2(float* __restrict__ out, const float* __restrict__ b2, int n) {
    int i = blockIdx.x * 256 + threadIdx.x;
    if (i < n) out[i] += b2[i % NCLS];
}

// ---------------------------- launch ----------------------------
extern "C" void kernel_launch(void* const* d_in, const int* in_sizes, int n_in,
                              void* d_out, int out_size, void* d_ws, size_t ws_size,
                              hipStream_t stream) {
    const float* x   = (const float*)d_in[0];
    const int*   ei  = (const int*)d_in[1];
    const float* W1  = (const float*)d_in[2];
    const float* a1s = (const float*)d_in[3];
    const float* a1d = (const float*)d_in[4];
    const float* b1  = (const float*)d_in[5];
    const float* W2  = (const float*)d_in[6];
    const float* a2s = (const float*)d_in[7];
    const float* a2d = (const float*)d_in[8];
    const float* b2  = (const float*)d_in[9];
    float* out = (float*)d_out;

    const int E    = in_sizes[1] / 2;
    const int Etot = E + NODES;
    const int* srcA = ei;
    const int* dstA = ei + E;

    // workspace carve (256B aligned regions)
    char* p = (char*)d_ws;
    auto carve = [&](size_t bytes) -> void* {
        void* r = (void*)p;
        p += (bytes + 255) & ~(size_t)255;
        return r;
    };
    half_t* xf16  = (half_t*)carve((size_t)NODES * FIN * sizeof(half_t));
    half_t* w1t   = (half_t*)carve((size_t)HC * FIN * sizeof(half_t));
    half_t* w2t   = (half_t*)carve((size_t)NCLSP * HC * sizeof(half_t));
    half_t* hr16  = (half_t*)carve((size_t)NODES * HC * sizeof(half_t));
    float*  h1    = (float*)carve((size_t)NODES * HC * sizeof(float));
    float*  agg1  = (float*)carve((size_t)NODES * HC * sizeof(float));
    float*  h2    = (float*)carve((size_t)NODES * NCLSP * sizeof(float));
    float*  als1  = (float*)carve((size_t)NODES * HEADS * sizeof(float));
    float*  ald1  = (float*)carve((size_t)NODES * HEADS * sizeof(float));
    unsigned* m1  = (unsigned*)carve((size_t)NODES * HEADS * sizeof(unsigned));
    float*  z1    = (float*)carve((size_t)NODES * HEADS * sizeof(float));
    float*  als2  = (float*)carve((size_t)NODES * sizeof(float));
    float*  ald2  = (float*)carve((size_t)NODES * sizeof(float));
    unsigned* m2  = (unsigned*)carve((size_t)NODES * sizeof(unsigned));
    float*  z2    = (float*)carve((size_t)NODES * sizeof(float));

    // zero-init accumulators (fenc init 0 < encoding of every finite float)
    hipMemsetAsync(m1,   0, (size_t)NODES * HEADS * 4, stream);
    hipMemsetAsync(z1,   0, (size_t)NODES * HEADS * 4, stream);
    hipMemsetAsync(agg1, 0, (size_t)NODES * HC * 4, stream);
    hipMemsetAsync(m2,   0, (size_t)NODES * 4, stream);
    hipMemsetAsync(z2,   0, (size_t)NODES * 4, stream);
    hipMemsetAsync(out,  0, (size_t)out_size * 4, stream);

    auto cdiv = [](long long a, long long b) { return (unsigned)((a + b - 1) / b); };

    // f16 conversions / transposes
    k_cvt_x  <<<cdiv((long long)NODES * FIN, 256), 256, 0, stream>>>(x, xf16, NODES * FIN);
    k_cvt_w1t<<<cdiv((long long)FIN * HC, 256),   256, 0, stream>>>(W1, w1t);
    k_cvt_w2t<<<cdiv((long long)NCLSP * HC, 256), 256, 0, stream>>>(W2, w2t);

    // layer 1
    k_gemm1  <<<NODES / 16, 256, 0, stream>>>(xf16, w1t, h1);
    k_attlog1<<<cdiv((long long)NODES * HEADS, 256), 256, 0, stream>>>(h1, a1s, a1d, als1, ald1);
    {
        int tot = Etot * HEADS;
        k_edge_max1<<<cdiv(tot, 256), 256, 0, stream>>>(srcA, dstA, E, tot, als1, ald1, m1);
        k_edge_sum1<<<cdiv(tot, 256), 256, 0, stream>>>(srcA, dstA, E, tot, als1, ald1, m1, z1);
    }
    k_scatter1<<<Etot, 256, 0, stream>>>(srcA, dstA, E, als1, ald1, m1, z1, h1, agg1);
    k_relu_cvt<<<cdiv((long long)NODES * HC, 256), 256, 0, stream>>>(agg1, b1, hr16);

    // layer 2
    k_gemm2  <<<cdiv(NODES, 128), 256, 0, stream>>>(hr16, w2t, h2);
    k_attlog2<<<cdiv(NODES, 256), 256, 0, stream>>>(h2, a2s, a2d, als2, ald2);
    k_edge_max2<<<cdiv(Etot, 256), 256, 0, stream>>>(srcA, dstA, E, Etot, als2, ald2, m2);
    k_edge_sum2<<<cdiv(Etot, 256), 256, 0, stream>>>(srcA, dstA, E, Etot, als2, ald2, m2, z2);
    {
        long long tot = (long long)Etot * NCLS;
        k_scatter2<<<cdiv(tot, 256), 256, 0, stream>>>(srcA, dstA, E, (int)tot, als2, ald2,
                                                       m2, z2, h2, out);
    }
    k_bias2<<<cdiv(out_size, 256), 256, 0, stream>>>(out, b2, out_size);
}
